// SchNetEmbedding_54503134986793
// MI455X (gfx1250) — compile-verified, matrix operand order checked by reading, not run on hardware
//
#include <hip/hip_runtime.h>
#include <hip/hip_bf16.h>
#include <math.h>

typedef __attribute__((ext_vector_type(16))) _Float16 v16h;
typedef __attribute__((ext_vector_type(8)))  float    v8f;
typedef unsigned int u32;
typedef __attribute__((ext_vector_type(4))) u32 u32x4;
typedef __attribute__((ext_vector_type(4))) int i32x4;
typedef __attribute__((ext_vector_type(8))) int i32x8;

#define NNODES 50000
#define NEDGES 800000
#define DIMC   128
#define NB     32
#define CUT    5.0f

#define WMMA(a, b, c) \
  __builtin_amdgcn_wmma_f32_16x16x32_f16(false, (a), false, (b), (short)0, (c), false, false)

#if defined(__has_builtin)
#if __has_builtin(__builtin_amdgcn_tensor_load_to_lds)
#define HAVE_TDM 1
#endif
#if __has_builtin(__builtin_amdgcn_s_wait_tensorcnt)
#define HAVE_WAIT_TENSOR 1
#endif
#endif

__device__ __forceinline__ void wait_tensorcnt0() {
#ifdef HAVE_WAIT_TENSOR
  __builtin_amdgcn_s_wait_tensorcnt(0);
#else
  asm volatile("s_wait_tensorcnt 0x0" ::: "memory");
#endif
}

// Issue one TDM DMA: nhalves f16 from global -> LDS (1-D tile, data_size=2B).
// Descriptor layout per CDNA5 ISA 8.3/8.4. Must be called by exactly one wave.
__device__ __forceinline__ void tdm_stage_f16(const _Float16* __restrict__ g,
                                              void* lds_dst, u32 nhalves) {
#ifdef HAVE_TDM
  unsigned long long ga = (unsigned long long)(const void*)g;
  u32 lds_off = (u32)(unsigned long long)lds_dst;  // addrspace(3) offset
  u32x4 g0;
  g0.x = 1u;                                        // count=1, user mode
  g0.y = lds_off;                                   // lds_addr [63:32]
  g0.z = (u32)ga;                                   // global_addr low
  g0.w = (u32)((ga >> 32) & 0x1FFFFFFu) | (2u << 30);  // addr[56:32] | type=2
  i32x8 g1;
  g1[0] = (int)(1u << 16);                          // data_size=1 (2 bytes)
  g1[1] = (int)((nhalves & 0xFFFFu) << 16);         // tensor_dim0[15:0]
  g1[2] = (int)((nhalves >> 16) | (1u << 16));      // tensor_dim0[31:16] | tensor_dim1=1
  g1[3] = (int)((nhalves & 0xFFFFu) << 16);         // tile_dim0 (<=65535)
  g1[4] = 1;                                        // tile_dim1=1, tile_dim2=0
  g1[5] = (int)nhalves;                             // tensor_dim0_stride[31:0]
  g1[6] = 0;
  g1[7] = 0;
  i32x4 z4 = {0, 0, 0, 0};
#if defined(__clang_major__) && (__clang_major__ >= 23)
  i32x8 z8 = {0, 0, 0, 0, 0, 0, 0, 0};
  __builtin_amdgcn_tensor_load_to_lds(g0, g1, z4, z4, z8, 0);
#else
  __builtin_amdgcn_tensor_load_to_lds(g0, g1, z4, z4, 0);
#endif
  wait_tensorcnt0();
#endif
}

// Cooperative fallback copy (also used when TDM builtin is unavailable).
__device__ __forceinline__ void coop_stage(const void* __restrict__ g, void* l,
                                           int nuint4) {
  const uint4* src = (const uint4*)g;
  uint4* dst = (uint4*)l;
  for (int i = threadIdx.x; i < nuint4; i += 256) dst[i] = src[i];
}

__device__ __forceinline__ float softplus_f(float x) {
  return (x > 20.0f) ? x : log1pf(__expf(x));
}

// k-offset inside a 16x32 f16 A/B-tile for packed element h (0..15) of a lane.
__device__ __forceinline__ int wmma_k16(int h, int hi) {
  int v = h >> 1, odd = h & 1;
  return (v < 4 ? 2 * v : 16 + 2 * (v - 4)) + odd + (hi ? 8 : 0);
}

// A-tile 16x32, rows base_m.., cols base_k.., from row-major f32 [*, ld]
__device__ __forceinline__ v16h load_a_f32(const float* __restrict__ X, int ld,
                                           int base_m, int base_k, int lane) {
  const int m = base_m + (lane & 15);
  const int hi = lane >> 4;
  v16h a;
#pragma unroll
  for (int h = 0; h < 16; ++h)
    a[h] = (_Float16)X[(size_t)m * ld + base_k + wmma_k16(h, hi)];
  return a;
}

// A-tile 16x32 from row-major f16 [*, ld]
__device__ __forceinline__ v16h load_a_f16(const _Float16* __restrict__ X, int ld,
                                           int base_m, int base_k, int lane) {
  const int m = base_m + (lane & 15);
  const int hi = lane >> 4;
  v16h a;
#pragma unroll
  for (int h = 0; h < 16; ++h)
    a[h] = X[(size_t)m * ld + base_k + wmma_k16(h, hi)];
  return a;
}

// ---------------- weight pre-pack ----------------
__global__ void __launch_bounds__(256) pack_w_kernel(const float* __restrict__ W,
                                                     _Float16* __restrict__ pk,
                                                     int ktiles) {
  int idx = blockIdx.x * blockDim.x + threadIdx.x;
  int total = ktiles * 8 * 32 * 16;
  if (idx >= total) return;
  int h = idx & 15;
  int lane = (idx >> 4) & 31;
  int t = idx >> 9;
  int nt = t & 7, kt = t >> 3;
  int hi = lane >> 4, n_lo = lane & 15;
  int k = kt * 32 + wmma_k16(h, hi);
  int n = nt * 16 + n_lo;
  pk[idx] = (_Float16)W[(size_t)k * DIMC + n];
}

// ---------------- elementwise kernels ----------------

__global__ void __launch_bounds__(256) rbf_kernel(const float* __restrict__ dist,
                                                  _Float16* __restrict__ rb, int total) {
  int i = blockIdx.x * blockDim.x + threadIdx.x;
  if (i >= total) return;
  int e = i >> 5, j = i & 31;
  float mu = (CUT / (NB - 1)) * (float)j;
  float sig = CUT / (float)NB;
  float t = (dist[e] - mu) / sig;
  rb[i] = (_Float16)__expf(-0.5f * t * t);
}

__global__ void __launch_bounds__(256) embed_kernel(const int* __restrict__ species,
                                                    const float* __restrict__ w,
                                                    const float* __restrict__ b,
                                                    _Float16* __restrict__ xi, int total) {
  int i = blockIdx.x * blockDim.x + threadIdx.x;
  if (i >= total) return;
  int node = i >> 7, c = i & 127;
  xi[i] = (_Float16)(w[(size_t)species[node] * DIMC + c] + b[c]);
}

// ---------------- node GEMM: xi(f16) = X @ W + b ----------------

__global__ void __launch_bounds__(256) node_gemm_kernel(const float* __restrict__ X,
                                                        const v16h* __restrict__ wpk,
                                                        const float* __restrict__ B,
                                                        _Float16* __restrict__ Y, int nrows) {
  __shared__ v16h s_w[1024];  // 32 KB
#ifdef HAVE_TDM
  if (threadIdx.x < 32) tdm_stage_f16((const _Float16*)wpk, s_w, DIMC * DIMC);
#else
  coop_stage(wpk, s_w, 2048);
#endif
  __syncthreads();

  const int lane = threadIdx.x & 31;
  const int wave = blockIdx.x * (blockDim.x >> 5) + (threadIdx.x >> 5);
  const int base_m = wave * 16;
  if (base_m >= nrows) return;
  const int n_lo = lane & 15, hi = lane >> 4;

  v16h a0 = load_a_f32(X, DIMC, base_m, 0, lane);
  v16h a1 = load_a_f32(X, DIMC, base_m, 32, lane);
  v16h a2 = load_a_f32(X, DIMC, base_m, 64, lane);
  v16h a3 = load_a_f32(X, DIMC, base_m, 96, lane);

#pragma unroll
  for (int nt = 0; nt < 8; ++nt) {
    v8f acc = {};
    acc = WMMA(a0, s_w[(0 * 8 + nt) * 32 + lane], acc);
    acc = WMMA(a1, s_w[(1 * 8 + nt) * 32 + lane], acc);
    acc = WMMA(a2, s_w[(2 * 8 + nt) * 32 + lane], acc);
    acc = WMMA(a3, s_w[(3 * 8 + nt) * 32 + lane], acc);
    const int n = nt * 16 + n_lo;
    const float bias = B[n];
#pragma unroll
    for (int r = 0; r < 8; ++r) {
      int m = base_m + hi * 8 + r;
      Y[(size_t)m * DIMC + n] = (_Float16)(acc[r] + bias);
    }
  }
}

// ---------------- fused edge kernel ----------------

__global__ void __launch_bounds__(256) edge_filter_message_kernel(
    const _Float16* __restrict__ rb,
    const v16h* __restrict__ f0pk, const float* __restrict__ f0b,
    const v16h* __restrict__ f1pk, const float* __restrict__ f1b,
    const _Float16* __restrict__ xi,
    const int* __restrict__ esrc, const int* __restrict__ edst,
    float* __restrict__ msg, int nedges) {
  __shared__ v16h s_f0[256];             // 8 KB
  __shared__ v16h s_f1[1024];            // 32 KB
  __shared__ _Float16 s_h[8][16][136];   // per-wave 16x128 f16, padded rows
#ifdef HAVE_TDM
  if (threadIdx.x < 32) {
    tdm_stage_f16((const _Float16*)f0pk, s_f0, NB * DIMC);
    tdm_stage_f16((const _Float16*)f1pk, s_f1, DIMC * DIMC);
  }
#else
  coop_stage(f0pk, s_f0, 512);
  coop_stage(f1pk, s_f1, 2048);
#endif
  __syncthreads();

  const int lane = threadIdx.x & 31;
  const int wslot = threadIdx.x >> 5;
  const int wave = blockIdx.x * 8 + wslot;
  const int e_base = wave * 16;
  if (e_base >= nedges) return;
  const int n_lo = lane & 15, hi = lane >> 4;

  // GEMM1 (K=32): h tile -> softplus -> LDS f16
  v16h a0 = load_a_f16(rb, NB, e_base, 0, lane);
#pragma unroll
  for (int nt = 0; nt < 8; ++nt) {
    v8f c = {};
    c = WMMA(a0, s_f0[nt * 32 + lane], c);
    const int n = nt * 16 + n_lo;
    const float bias = f0b[n];
#pragma unroll
    for (int r = 0; r < 8; ++r)
      s_h[wslot][hi * 8 + r][n] = (_Float16)softplus_f(c[r] + bias);
  }
  asm volatile("s_wait_dscnt 0x0" ::: "memory");

  // GEMM2 (K=128): w = h @ f1w
  v8f acc[8];
  const v8f vzero = {};
#pragma unroll
  for (int nt = 0; nt < 8; ++nt) acc[nt] = vzero;
#pragma unroll
  for (int kt = 0; kt < 4; ++kt) {
    v16h a = load_a_f16(&s_h[wslot][0][0], 136, 0, kt * 32, lane);
#pragma unroll
    for (int nt = 0; nt < 8; ++nt)
      acc[nt] = WMMA(a, s_f1[(kt * 8 + nt) * 32 + lane], acc[nt]);
  }

  // gather node features, multiply, scatter with run-merged atomics
  int src[8], dst[8];
#pragma unroll
  for (int r = 0; r < 8; ++r) {
    int e = e_base + hi * 8 + r;
    src[r] = esrc[e];
    dst[r] = edst[e];
  }
#pragma unroll
  for (int nt = 0; nt < 8; ++nt) {
    const int n = nt * 16 + n_lo;
    const float bias = f1b[n];
    float pending = 0.0f;
    int cur = src[0];
#pragma unroll
    for (int r = 0; r < 8; ++r) {
      float wv = acc[nt][r] + bias;
      float xv = (float)xi[(size_t)dst[r] * DIMC + n];
      if (src[r] != cur) {
        atomicAdd(&msg[(size_t)cur * DIMC + n], pending);
        pending = 0.0f;
        cur = src[r];
      }
      pending += wv * xv;
    }
    atomicAdd(&msg[(size_t)cur * DIMC + n], pending);
  }
}

// ---------------- fused node output kernel ----------------

__global__ void __launch_bounds__(256) node_out_kernel(
    const float* __restrict__ msg,
    const v16h* __restrict__ w2pk, const float* __restrict__ b2,
    const v16h* __restrict__ w3pk, const float* __restrict__ b3,
    const float* __restrict__ xprev, int add_residual,
    float* __restrict__ out, int nrows) {
  __shared__ v16h s_w2[1024];
  __shared__ v16h s_w3[1024];
  __shared__ _Float16 s_h[8][16][136];
#ifdef HAVE_TDM
  if (threadIdx.x < 32) {
    tdm_stage_f16((const _Float16*)w2pk, s_w2, DIMC * DIMC);
    tdm_stage_f16((const _Float16*)w3pk, s_w3, DIMC * DIMC);
  }
#else
  coop_stage(w2pk, s_w2, 2048);
  coop_stage(w3pk, s_w3, 2048);
#endif
  __syncthreads();

  const int lane = threadIdx.x & 31;
  const int wslot = threadIdx.x >> 5;
  const int wave = blockIdx.x * 8 + wslot;
  const int base_m = wave * 16;
  if (base_m >= nrows) return;
  const int n_lo = lane & 15, hi = lane >> 4;

  // GEMM1 (K=128): t = softplus(msg@w2+b2) -> LDS f16
  v16h a0 = load_a_f32(msg, DIMC, base_m, 0, lane);
  v16h a1 = load_a_f32(msg, DIMC, base_m, 32, lane);
  v16h a2 = load_a_f32(msg, DIMC, base_m, 64, lane);
  v16h a3 = load_a_f32(msg, DIMC, base_m, 96, lane);
#pragma unroll
  for (int nt = 0; nt < 8; ++nt) {
    v8f c = {};
    c = WMMA(a0, s_w2[(0 * 8 + nt) * 32 + lane], c);
    c = WMMA(a1, s_w2[(1 * 8 + nt) * 32 + lane], c);
    c = WMMA(a2, s_w2[(2 * 8 + nt) * 32 + lane], c);
    c = WMMA(a3, s_w2[(3 * 8 + nt) * 32 + lane], c);
    const int n = nt * 16 + n_lo;
    const float bias = b2[n];
#pragma unroll
    for (int r = 0; r < 8; ++r)
      s_h[wslot][hi * 8 + r][n] = (_Float16)softplus_f(c[r] + bias);
  }
  asm volatile("s_wait_dscnt 0x0" ::: "memory");

  // GEMM2 (K=128): out = t @ w3 + b3 (+ residual)
  v8f acc[8];
  const v8f vzero = {};
#pragma unroll
  for (int nt = 0; nt < 8; ++nt) acc[nt] = vzero;
#pragma unroll
  for (int kt = 0; kt < 4; ++kt) {
    v16h a = load_a_f16(&s_h[wslot][0][0], 136, 0, kt * 32, lane);
#pragma unroll
    for (int nt = 0; nt < 8; ++nt)
      acc[nt] = WMMA(a, s_w3[(kt * 8 + nt) * 32 + lane], acc[nt]);
  }
#pragma unroll
  for (int nt = 0; nt < 8; ++nt) {
    const int n = nt * 16 + n_lo;
    const float bias = b3[n];
#pragma unroll
    for (int r = 0; r < 8; ++r) {
      size_t idx = (size_t)(base_m + hi * 8 + r) * DIMC + n;
      float v = acc[nt][r] + bias;
      if (add_residual) v += xprev[idx];
      out[idx] = v;
    }
  }
}

// ---------------- host launch ----------------

extern "C" void kernel_launch(void* const* d_in, const int* in_sizes, int n_in,
                              void* d_out, int out_size, void* d_ws, size_t ws_size,
                              hipStream_t stream) {
  const int*   species = (const int*)d_in[0];
  const int*   esrc    = (const int*)d_in[1];
  const int*   edst    = (const int*)d_in[2];
  const float* dist    = (const float*)d_in[3];
  const float* aw1w    = (const float*)d_in[4];
  const float* aw1b    = (const float*)d_in[5];
  const float* aw2w    = (const float*)d_in[6];
  const float* aw2b    = (const float*)d_in[7];
  const float* aw3w    = (const float*)d_in[8];
  const float* aw3b    = (const float*)d_in[9];
  const float* f0w     = (const float*)d_in[10];
  const float* f0b     = (const float*)d_in[11];
  const float* f1w     = (const float*)d_in[12];
  const float* f1b     = (const float*)d_in[13];

  char* ws = (char*)d_ws;
  _Float16* rb   = (_Float16*)ws;  ws += (size_t)NEDGES * NB * sizeof(_Float16);
  _Float16* xi   = (_Float16*)ws;  ws += (size_t)NNODES * DIMC * sizeof(_Float16);
  float* msg     = (float*)ws;     ws += (size_t)NNODES * DIMC * sizeof(float);
  float* xbuf    = (float*)ws;     ws += (size_t)NNODES * DIMC * sizeof(float);
  _Float16* w1pk = (_Float16*)ws;  ws += (size_t)3 * DIMC * DIMC * sizeof(_Float16);
  _Float16* w2pk = (_Float16*)ws;  ws += (size_t)3 * DIMC * DIMC * sizeof(_Float16);
  _Float16* w3pk = (_Float16*)ws;  ws += (size_t)3 * DIMC * DIMC * sizeof(_Float16);
  _Float16* f1pk = (_Float16*)ws;  ws += (size_t)3 * DIMC * DIMC * sizeof(_Float16);
  _Float16* f0pk = (_Float16*)ws;  ws += (size_t)3 * NB * DIMC * sizeof(_Float16);

  rbf_kernel<<<(NEDGES * NB + 255) / 256, 256, 0, stream>>>(dist, rb, NEDGES * NB);
  for (int l = 0; l < 3; ++l) {
    size_t o128 = (size_t)l * DIMC * DIMC;
    size_t o32  = (size_t)l * NB * DIMC;
    pack_w_kernel<<<64, 256, 0, stream>>>(aw1w + o128, w1pk + o128, 4);
    pack_w_kernel<<<64, 256, 0, stream>>>(aw2w + o128, w2pk + o128, 4);
    pack_w_kernel<<<64, 256, 0, stream>>>(aw3w + o128, w3pk + o128, 4);
    pack_w_kernel<<<64, 256, 0, stream>>>(f1w + o128, f1pk + o128, 4);
    pack_w_kernel<<<16, 256, 0, stream>>>(f0w + o32, f0pk + o32, 1);
  }

  const int node_waves  = NNODES / 16;                  // 3125 (exact)
  const int node_blocks = (node_waves + 7) / 8;
  const int edge_blocks = (NEDGES / 16 + 7) / 8;        // exact

  for (int l = 0; l < 3; ++l) {
    size_t o128 = (size_t)l * DIMC * DIMC;
    size_t o32  = (size_t)l * NB * DIMC;
    const float* b1 = aw1b + (size_t)l * DIMC;
    const float* b2 = aw2b + (size_t)l * DIMC;
    const float* b3 = aw3b + (size_t)l * DIMC;
    const float* bf0 = f0b + (size_t)l * DIMC;
    const float* bf1 = f1b + (size_t)l * DIMC;

    if (l == 0)
      embed_kernel<<<(NNODES * DIMC + 255) / 256, 256, 0, stream>>>(
          species, aw1w, b1, xi, NNODES * DIMC);
    else
      node_gemm_kernel<<<node_blocks, 256, 0, stream>>>(
          xbuf, (const v16h*)(w1pk + o128), b1, xi, NNODES);

    hipMemsetAsync(msg, 0, (size_t)NNODES * DIMC * sizeof(float), stream);

    edge_filter_message_kernel<<<edge_blocks, 256, 0, stream>>>(
        rb, (const v16h*)(f0pk + o32), bf0, (const v16h*)(f1pk + o128), bf1,
        xi, esrc, edst, msg, NEDGES);

    float* outp = (l == 2) ? (float*)d_out : xbuf;
    node_out_kernel<<<node_blocks, 256, 0, stream>>>(
        msg, (const v16h*)(w2pk + o128), b2, (const v16h*)(w3pk + o128), b3,
        xbuf, (l > 0) ? 1 : 0, outp, NNODES);
  }
}
